// UniversalBehavioralTransformer_33054068310700
// MI455X (gfx1250) — compile-verified, matrix operand order checked by reading, not run on hardware
//
#include <hip/hip_runtime.h>
#include <hip/hip_bf16.h>
#include <math.h>

// ---------------------------------------------------------------------------
// Problem constants (B,N,D)=(8,1024,1024); H=8, DQK=DV=128, L=2, TB=128
// ---------------------------------------------------------------------------
constexpr int kB = 8, kN = 1024, kD = 1024, kH = 8, kDQK = 128, kDV = 128;
constexpr int kL = 2, kTB = 128;
constexpr int kM = kB * kN;          // 8192 tokens
constexpr int kNuvqk = 4096;         // 2*H*DV + 2*H*DQK

typedef __bf16 bf16;
typedef __attribute__((ext_vector_type(16))) __bf16 v16bf;
typedef __attribute__((ext_vector_type(8)))  __bf16 v8bf;
typedef __attribute__((ext_vector_type(8)))  float  v8f;

__device__ __forceinline__ v8f zero_v8f() {
  v8f z;
#pragma unroll
  for (int i = 0; i < 8; ++i) z[i] = 0.0f;
  return z;
}

__device__ __forceinline__ v8f wmma_bf16(v16bf a, v16bf b, v8f c) {
  // D = A(16x32) * B(32x16) + C(16x16 f32)
  return __builtin_amdgcn_wmma_f32_16x16x32_bf16(false, a, false, b,
                                                 (short)0, c, false, false);
}

// Load a 16x32 bf16 WMMA operand fragment. `base` points at [row0][k0] of a
// row-major (k-contiguous) matrix with row stride `stride` elements.
// ISA layout: lane L -> row = L&15, kb = (L>>4)*8, k in {kb..kb+7, kb+16..kb+23}
__device__ __forceinline__ v16bf load_frag(const bf16* __restrict__ base,
                                           int stride, int lane) {
  const int r  = lane & 15;
  const int kb = (lane >> 4) << 3;
  const bf16* p = base + (size_t)r * stride + kb;
  v8bf lo = *(const v8bf*)(p);
  v8bf hi = *(const v8bf*)(p + 16);
  v16bf out;
#pragma unroll
  for (int i = 0; i < 8; ++i) { out[i] = lo[i]; out[i + 8] = hi[i]; }
  return out;
}

__device__ __forceinline__ float block_reduce_sum(float v, float* red) {
  const int t = threadIdx.x;
  red[t] = v;
  __syncthreads();
  for (int s = 128; s > 0; s >>= 1) {
    if (t < s) red[t] += red[t + s];
    __syncthreads();
  }
  float r = red[0];
  __syncthreads();
  return r;
}

// Fast silu: single v_exp_f32 + single v_rcp_f32 (TRANS ops co-execute with
// WMMA on CDNA5); avoids the IEEE div_scale/div_fmas refinement sequence.
__device__ __forceinline__ float silu_f(float v) {
  return v * __builtin_amdgcn_rcpf(1.0f + __expf(-v));
}

// ---------------------------------------------------------------------------
// Prep kernels
// ---------------------------------------------------------------------------
__global__ void prep_x_kernel(const float* __restrict__ x,
                              const float* __restrict__ mask,
                              float* __restrict__ xw) {
  size_t tid = (size_t)blockIdx.x * blockDim.x + threadIdx.x;
  xw[tid] = x[tid] * mask[tid >> 10];  // D == 1024
}

// src: [L*K][C] f32 row-major  ->  dst: [L][C][K] bf16 (k-contiguous)
// LDS-tiled 32x32 transpose; block (32,8); grid (C/32, L*K/32).
__global__ __launch_bounds__(256) void transpose_w_kernel(
    const float* __restrict__ src, bf16* __restrict__ dst, int K, int C) {
  __shared__ bf16 tile[32][33];
  const int tx = threadIdx.x, ty = threadIdx.y;
  const int c0 = blockIdx.x * 32;
  const int r0 = blockIdx.y * 32;  // global row in [0, L*K)
#pragma unroll
  for (int i = 0; i < 4; ++i) {
    const int rr = ty + 8 * i;
    tile[rr][tx] = (bf16)src[(size_t)(r0 + rr) * C + c0 + tx];
  }
  __syncthreads();
  const int l = r0 / K;          // K multiple of 32 -> block never straddles l
  const int kbase = r0 - l * K;
#pragma unroll
  for (int i = 0; i < 4; ++i) {
    const int c = c0 + ty + 8 * i;
    dst[((size_t)l * C + c) * K + kbase + tx] = tile[tx][ty + 8 * i];
  }
}

// rab[b,n,m] = pos_w[n-m+N-1] + ts_w[clip(trunc(log|dt|/log2),0,TB)]
__global__ void rab_kernel(const int* __restrict__ ts,
                           const float* __restrict__ pos_w,
                           const float* __restrict__ ts_w,
                           float* __restrict__ rab) {
  const int n = blockIdx.x, b = blockIdx.y;
  const int np1 = (n + 1 < kN) ? (n + 1) : (kN - 1);
  const long long tsn = (long long)ts[b * kN + np1];
  for (int m = threadIdx.x; m < kN; m += blockDim.x) {
    long long delta = tsn - (long long)ts[b * kN + m];
    float ad = fabsf((float)delta);
    ad = fmaxf(ad, 1.0f);
    int bucket = (int)(__logf(ad) * 1.4426950408889634f);
    bucket = bucket < 0 ? 0 : (bucket > kTB ? kTB : bucket);
    rab[((size_t)b * kN + n) * kN + m] = pos_w[n - m + kN - 1] + ts_w[bucket];
  }
}

// ---------------------------------------------------------------------------
// LayerNorm kernels (one block of 256 per token, D=1024)
// ---------------------------------------------------------------------------
__global__ __launch_bounds__(256) void ln1_kernel(const float* __restrict__ x,
                                                  const float* __restrict__ g,
                                                  const float* __restrict__ bt,
                                                  bf16* __restrict__ out) {
  __shared__ float red[256];
  const size_t t = blockIdx.x;
  const float* xr = x + t * kD;
  float v[4]; float s = 0.0f;
#pragma unroll
  for (int i = 0; i < 4; ++i) { v[i] = xr[threadIdx.x + i * 256]; s += v[i]; }
  const float mu = block_reduce_sum(s, red) * (1.0f / (float)kD);
  float s2 = 0.0f;
#pragma unroll
  for (int i = 0; i < 4; ++i) { float d = v[i] - mu; s2 += d * d; }
  const float var = block_reduce_sum(s2, red) * (1.0f / (float)kD);
  const float rs = rsqrtf(var + 1e-6f);
#pragma unroll
  for (int i = 0; i < 4; ++i) {
    int d = threadIdx.x + i * 256;
    out[t * kD + d] = (bf16)((v[i] - mu) * rs * g[d] + bt[d]);
  }
}

// g_bf = bf16( (LN2(o)*g2+b2) * u )   (1024 wide)
__global__ __launch_bounds__(256) void ln2u_kernel(const float* __restrict__ o,
                                                   const float* __restrict__ g,
                                                   const float* __restrict__ bt,
                                                   const float* __restrict__ u,
                                                   bf16* __restrict__ out) {
  __shared__ float red[256];
  const size_t t = blockIdx.x;
  const float* xr = o + t * kD;
  float v[4]; float s = 0.0f;
#pragma unroll
  for (int i = 0; i < 4; ++i) { v[i] = xr[threadIdx.x + i * 256]; s += v[i]; }
  const float mu = block_reduce_sum(s, red) * (1.0f / (float)kD);
  float s2 = 0.0f;
#pragma unroll
  for (int i = 0; i < 4; ++i) { float d = v[i] - mu; s2 += d * d; }
  const float var = block_reduce_sum(s2, red) * (1.0f / (float)kD);
  const float rs = rsqrtf(var + 1e-6f);
#pragma unroll
  for (int i = 0; i < 4; ++i) {
    int d = threadIdx.x + i * 256;
    float val = ((v[i] - mu) * rs * g[d] + bt[d]) * u[t * kD + d];
    out[t * kD + d] = (bf16)val;
  }
}

// ---------------------------------------------------------------------------
// GEMM 1: uvqk = silu(h @ Wt^T + bias); scatter epilogue
//   A: h_bf [8192][1024]; Bt: [4096][1024] (k-contiguous per output col)
//   8 waves/block as 2(M) x 4(N); wave tile 64x64 -> block tile 128x256
//   Per K-step: 4 A-frags + 4 B-frags (16 b128 loads) feed 16 WMMA.
// ---------------------------------------------------------------------------
__global__ __launch_bounds__(256) void gemm_uvqk_kernel(
    const bf16* __restrict__ A, const bf16* __restrict__ Bt,
    const float* __restrict__ bias, float* __restrict__ u_out,
    bf16* __restrict__ q_out, bf16* __restrict__ k_out,
    bf16* __restrict__ vT_out) {
  const int lane = threadIdx.x & 31;
  const int wave = threadIdx.x >> 5;
  const int row0 = blockIdx.y * 128 + (wave >> 2) * 64;
  const int col0 = blockIdx.x * 256 + (wave & 3) * 64;
  v8f acc[4][4];
#pragma unroll
  for (int i = 0; i < 4; ++i)
#pragma unroll
    for (int j = 0; j < 4; ++j) acc[i][j] = zero_v8f();

  for (int k0 = 0; k0 < kD; k0 += 32) {
    v16bf a[4], b[4];
#pragma unroll
    for (int i = 0; i < 4; ++i)
      a[i] = load_frag(A + (size_t)(row0 + i * 16) * kD + k0, kD, lane);
#pragma unroll
    for (int j = 0; j < 4; ++j)
      b[j] = load_frag(Bt + (size_t)(col0 + j * 16) * kD + k0, kD, lane);
#pragma unroll
    for (int i = 0; i < 4; ++i)
#pragma unroll
      for (int j = 0; j < 4; ++j) acc[i][j] = wmma_bf16(a[i], b[j], acc[i][j]);
  }

  const int cl = lane & 15;
  const int rsel = (lane >> 4) << 3;
#pragma unroll
  for (int i = 0; i < 4; ++i) {
#pragma unroll
    for (int j = 0; j < 4; ++j) {
#pragma unroll
      for (int e = 0; e < 8; ++e) {
        const int r = row0 + i * 16 + rsel + e;
        const int c = col0 + j * 16 + cl;
        float v = silu_f(acc[i][j][e] + bias[c]);
        const int bb = r >> 10, n = r & (kN - 1);
        if (c < 1024) {
          u_out[(size_t)r * 1024 + c] = v;
        } else if (c < 2048) {
          const int c2 = c - 1024, hh = c2 >> 7, d = c2 & 127;
          vT_out[(((size_t)bb * kH + hh) * kDV + d) * kN + n] = (bf16)v;
        } else if (c < 3072) {
          const int c2 = c - 2048, hh = c2 >> 7, d = c2 & 127;
          q_out[(((size_t)bb * kH + hh) * kN + n) * kDQK + d] = (bf16)v;
        } else {
          const int c2 = c - 3072, hh = c2 >> 7, d = c2 & 127;
          k_out[(((size_t)bb * kH + hh) * kN + n) * kDQK + d] = (bf16)v;
        }
      }
    }
  }
}

// ---------------------------------------------------------------------------
// GEMM 2: x += g @ WoT^T + b_o  (M=8192, N=1024, K=1024); same tiling
// ---------------------------------------------------------------------------
__global__ __launch_bounds__(256) void gemm_o_kernel(
    const bf16* __restrict__ A, const bf16* __restrict__ Bt,
    const float* __restrict__ bias, float* __restrict__ xio) {
  const int lane = threadIdx.x & 31;
  const int wave = threadIdx.x >> 5;
  const int row0 = blockIdx.y * 128 + (wave >> 2) * 64;
  const int col0 = blockIdx.x * 256 + (wave & 3) * 64;
  v8f acc[4][4];
#pragma unroll
  for (int i = 0; i < 4; ++i)
#pragma unroll
    for (int j = 0; j < 4; ++j) acc[i][j] = zero_v8f();

  for (int k0 = 0; k0 < kD; k0 += 32) {
    v16bf a[4], b[4];
#pragma unroll
    for (int i = 0; i < 4; ++i)
      a[i] = load_frag(A + (size_t)(row0 + i * 16) * kD + k0, kD, lane);
#pragma unroll
    for (int j = 0; j < 4; ++j)
      b[j] = load_frag(Bt + (size_t)(col0 + j * 16) * kD + k0, kD, lane);
#pragma unroll
    for (int i = 0; i < 4; ++i)
#pragma unroll
      for (int j = 0; j < 4; ++j) acc[i][j] = wmma_bf16(a[i], b[j], acc[i][j]);
  }

  const int cl = lane & 15;
  const int rsel = (lane >> 4) << 3;
#pragma unroll
  for (int i = 0; i < 4; ++i) {
#pragma unroll
    for (int j = 0; j < 4; ++j) {
#pragma unroll
      for (int e = 0; e < 8; ++e) {
        const int r = row0 + i * 16 + rsel + e;
        const int c = col0 + j * 16 + cl;
        xio[(size_t)r * kD + c] += acc[i][j][e] + bias[c];
      }
    }
  }
}

// ---------------------------------------------------------------------------
// Attention: o[n,:] = sum_m silu(q.k^T + rab)/N * amask * v
//   grid(N/64, H, B), 128 threads (4 waves x 16 q-rows). No softmax.
// ---------------------------------------------------------------------------
__global__ __launch_bounds__(128) void attn_kernel(
    const bf16* __restrict__ q, const bf16* __restrict__ k,
    const bf16* __restrict__ vT, const float* __restrict__ rab,
    const float* __restrict__ mask, float* __restrict__ o) {
  const int lane = threadIdx.x & 31;
  const int wave = threadIdx.x >> 5;  // 0..3
  const int bb = blockIdx.z, hh = blockIdx.y;
  const int n0 = blockIdx.x * 64 + wave * 16;
  const bf16* qb = q + ((size_t)bb * kH + hh) * kN * kDQK;   // [n][d]
  const bf16* kb = k + ((size_t)bb * kH + hh) * kN * kDQK;   // [m][d]
  const bf16* vb = vT + ((size_t)bb * kH + hh) * kDV * kN;   // [d][m]
  __shared__ __align__(16) bf16 smem[4][16][32];             // per-wave S tile

  v16bf qf[4];
#pragma unroll
  for (int kk = 0; kk < 4; ++kk)
    qf[kk] = load_frag(qb + (size_t)n0 * kDQK + kk * 32, kDQK, lane);

  v8f oacc[8];
#pragma unroll
  for (int dt = 0; dt < 8; ++dt) oacc[dt] = zero_v8f();

  const int cl = lane & 15;
  const int rsel = (lane >> 4) << 3;
  const int mc_limit = (blockIdx.x * 64 + 63) >> 5;  // uniform per block

  for (int mc = 0; mc <= mc_limit; ++mc) {
    const int mbase = mc * 32;
    const bool active = (mbase <= n0 + 15);  // uniform per wave
    if (active) {
#pragma unroll
      for (int sub = 0; sub < 2; ++sub) {
        const int m0 = mbase + sub * 16;
        v8f s = zero_v8f();
        if (m0 <= n0 + 15) {
#pragma unroll
          for (int kk = 0; kk < 4; ++kk)
            s = wmma_bf16(qf[kk],
                          load_frag(kb + (size_t)m0 * kDQK + kk * 32, kDQK, lane),
                          s);
        }
#pragma unroll
        for (int e = 0; e < 8; ++e) {
          const int r = rsel + e;
          const int n = n0 + r;
          const int m = m0 + cl;
          float val = 0.0f;
          if (m <= n) {
            float sv = s[e] + rab[((size_t)bb * kN + n) * kN + m];
            sv = silu_f(sv);
            val = sv * (1.0f / (float)kN) * mask[bb * kN + m];
          }
          smem[wave][r][sub * 16 + cl] = (bf16)val;
        }
      }
    }
    __syncthreads();
    if (active) {
      v16bf sf = load_frag(&smem[wave][0][0], 32, lane);  // S as A-fragment
#pragma unroll
      for (int dt = 0; dt < 8; ++dt) {
        v16bf vf = load_frag(vb + (size_t)(dt * 16) * kN + mbase, kN, lane);
        oacc[dt] = wmma_bf16(sf, vf, oacc[dt]);
      }
    }
    __syncthreads();
  }

#pragma unroll
  for (int dt = 0; dt < 8; ++dt) {
#pragma unroll
    for (int e = 0; e < 8; ++e) {
      const int n = n0 + rsel + e;
      const int d = dt * 16 + cl;
      o[((size_t)bb * kN + n) * (kH * kDV) + hh * kDV + d] = oacc[dt][e];
    }
  }
}

// ---------------------------------------------------------------------------
// Final pooling
// ---------------------------------------------------------------------------
__global__ __launch_bounds__(256) void norm_kernel(const float* __restrict__ x,
                                                   float* __restrict__ norms) {
  __shared__ float red[256];
  const size_t t = blockIdx.x;
  float s = 0.0f;
#pragma unroll
  for (int i = 0; i < 4; ++i) {
    float v = x[t * kD + threadIdx.x + i * 256];
    s += v * v;
  }
  const float ss = block_reduce_sum(s, red);
  if (threadIdx.x == 0) norms[t] = fmaxf(sqrtf(ss), 1e-12f);
}

__global__ __launch_bounds__(256) void pool_kernel(
    const float* __restrict__ x, const float* __restrict__ norms,
    const float* __restrict__ mask, float* __restrict__ out) {
  __shared__ float red[256];
  __shared__ float pooled[kD];
  const int bb = blockIdx.x;
  float ls = 0.0f;
  for (int n = threadIdx.x; n < kN; n += 256) ls += mask[bb * kN + n];
  const float len = block_reduce_sum(ls, red);
  const int last = (int)(len + 0.5f) - 1;

  float acc[4] = {0.0f, 0.0f, 0.0f, 0.0f};
  for (int n = 0; n < kN; ++n) {
    const float mk = mask[bb * kN + n];
    if (mk != 0.0f) {
      const float inv = 1.0f / norms[bb * kN + n];
      const float* xr = x + ((size_t)bb * kN + n) * kD;
#pragma unroll
      for (int i = 0; i < 4; ++i) acc[i] += xr[threadIdx.x + i * 256] * inv;
    }
  }
  const float invl = 1.0f / norms[bb * kN + last];
  const float* xl = x + ((size_t)bb * kN + last) * kD;
#pragma unroll
  for (int i = 0; i < 4; ++i) {
    const int d = threadIdx.x + i * 256;
    pooled[d] = 0.5f * (xl[d] * invl + acc[i] / len);
  }
  __syncthreads();
  float ss = 0.0f;
#pragma unroll
  for (int i = 0; i < 4; ++i) {
    const int d = threadIdx.x + i * 256;
    ss += pooled[d] * pooled[d];
  }
  const float tot = block_reduce_sum(ss, red);
  const float nrm = fmaxf(sqrtf(tot), 1e-12f);
#pragma unroll
  for (int i = 0; i < 4; ++i) {
    const int d = threadIdx.x + i * 256;
    out[bb * kD + d] = pooled[d] / nrm;
  }
}

// ---------------------------------------------------------------------------
// Host-side launch
// ---------------------------------------------------------------------------
extern "C" void kernel_launch(void* const* d_in, const int* in_sizes, int n_in,
                              void* d_out, int out_size, void* d_ws,
                              size_t ws_size, hipStream_t stream) {
  const float* x      = (const float*)d_in[0];
  const int*   ts     = (const int*)d_in[1];
  const float* mask   = (const float*)d_in[2];
  const float* ln1_g  = (const float*)d_in[3];
  const float* ln1_b  = (const float*)d_in[4];
  const float* w_uvqk = (const float*)d_in[5];
  const float* b_uvqk = (const float*)d_in[6];
  const float* ln2_g  = (const float*)d_in[7];
  const float* ln2_b  = (const float*)d_in[8];
  const float* w_o    = (const float*)d_in[9];
  const float* b_o    = (const float*)d_in[10];
  const float* pos_w  = (const float*)d_in[11];
  const float* ts_w   = (const float*)d_in[12];
  float* out = (float*)d_out;

  char* p = (char*)d_ws;
  auto take = [&](size_t bytes) -> char* {
    char* r = p;
    p += (bytes + 255) & ~((size_t)255);
    return r;
  };
  float* x_work = (float*)take((size_t)kM * kD * 4);          // 32 MB
  bf16*  h_bf   = (bf16*)take((size_t)kM * kD * 2);           // 16 MB
  float* u_f32  = (float*)take((size_t)kM * kD * 4);          // 32 MB
  bf16*  q_bf   = (bf16*)take((size_t)kB * kH * kN * kDQK * 2);
  bf16*  k_bf   = (bf16*)take((size_t)kB * kH * kN * kDQK * 2);
  bf16*  vT_bf  = (bf16*)take((size_t)kB * kH * kDV * kN * 2);
  float* o_f32  = (float*)take((size_t)kM * kD * 4);          // 32 MB
  bf16*  g_bf   = (bf16*)take((size_t)kM * kD * 2);           // 16 MB
  float* rab    = (float*)take((size_t)kB * kN * kN * 4);     // 32 MB
  bf16*  wuvqkT = (bf16*)take((size_t)kL * kNuvqk * kD * 2);  // 16 MB
  bf16*  woT    = (bf16*)take((size_t)kL * kD * kD * 2);      // 4 MB
  float* norms  = (float*)take((size_t)kM * 4);

  // Prep
  prep_x_kernel<<<(kM * kD) / 256, 256, 0, stream>>>(x, mask, x_work);
  transpose_w_kernel<<<dim3(kNuvqk / 32, (kL * kD) / 32), dim3(32, 8), 0,
                       stream>>>(w_uvqk, wuvqkT, kD, kNuvqk);
  transpose_w_kernel<<<dim3(kD / 32, (kL * kD) / 32), dim3(32, 8), 0,
                       stream>>>(w_o, woT, kD, kD);
  rab_kernel<<<dim3(kN, kB), 256, 0, stream>>>(ts, pos_w, ts_w, rab);

  for (int l = 0; l < kL; ++l) {
    ln1_kernel<<<kM, 256, 0, stream>>>(x_work, ln1_g + l * kD, ln1_b + l * kD,
                                       h_bf);
    gemm_uvqk_kernel<<<dim3(kNuvqk / 256, kM / 128), 256, 0, stream>>>(
        h_bf, wuvqkT + (size_t)l * kNuvqk * kD, b_uvqk + l * kNuvqk, u_f32,
        q_bf, k_bf, vT_bf);
    attn_kernel<<<dim3(kN / 64, kH, kB), 128, 0, stream>>>(q_bf, k_bf, vT_bf,
                                                           rab, mask, o_f32);
    ln2u_kernel<<<kM, 256, 0, stream>>>(o_f32, ln2_g + l * kD, ln2_b + l * kD,
                                        u_f32, g_bf);
    gemm_o_kernel<<<dim3(kD / 256, kM / 128), 256, 0, stream>>>(
        g_bf, woT + (size_t)l * kD * kD, b_o + l * kD, x_work);
  }

  norm_kernel<<<kM, 256, 0, stream>>>(x_work, norms);
  pool_kernel<<<kB, 256, 0, stream>>>(x_work, norms, mask, out);
  (void)in_sizes; (void)n_in; (void)out_size; (void)ws_size;
}
